// Attention_spatial_78778290144032
// MI455X (gfx1250) — matrix operand order repaired and checked by reading, not compiled, not run on hardware
//
#include <hip/hip_runtime.h>
#include <hip/hip_bf16.h>

// ---------------------------------------------------------------------------
// Spatial attention block (GroupNorm -> QKV 1x1 -> full spatial attention ->
// out 1x1 + residual) for B=2, C=256, H=W=64, 4 heads, head_dim=64.
// All matmuls on v_wmma_f32_16x16x32_f16 (gfx1250, wave32). Flash attention
// with block-shared K/V tiles staged in LDS; all WMMA operands are b128 runs.
// ---------------------------------------------------------------------------

typedef __attribute__((ext_vector_type(16))) _Float16 v16h;
typedef __attribute__((ext_vector_type(8)))  _Float16 v8h;
typedef __attribute__((ext_vector_type(4)))  _Float16 v4h;
typedef __attribute__((ext_vector_type(8)))  float    v8f;
typedef __attribute__((ext_vector_type(4)))  unsigned int u32x4;
typedef __attribute__((ext_vector_type(4)))  float    f32x4;

#define C_DIM   256
#define L_DIM   4096          // 64*64
#define B_DIM   2
#define NHEAD   4
#define HDIM    64
#define GROUPS  16
#define CPG     16            // channels per group

__device__ __forceinline__ v8f wmma16(v16h a, v16h b, v8f c) {
    return __builtin_amdgcn_wmma_f32_16x16x32_f16(
        false, a, false, b, (short)0, c, false, false);
}

// B-operand (K x 16, 16-bit): lane holds 16 contiguous K-halves -> 2 x b128
__device__ __forceinline__ v16h load16h(const _Float16* p) {
    union { u32x4 q[2]; v16h h; } u;
    u.q[0] = *(const u32x4*)p;
    u.q[1] = *(const u32x4*)(p + 8);
    return u.h;
}

// A-operand (16 x 32, 16-bit): elements 0..15 <- row[16*hi ..]
// (contiguous K, matching the B operand) -> 2 x b128
__device__ __forceinline__ v16h load_a16(const _Float16* row, int hi) {
    union { u32x4 q[2]; v16h h; } u;
    u.q[0] = *(const u32x4*)(row + 16 * hi);
    u.q[1] = *(const u32x4*)(row + 16 * hi + 8);
    return u.h;
}

// ---------------------------------------------------------------------------
// 0) One-shot weight conversion fp32 -> f16 (wqkv 768x256, wout 256x256).
// ---------------------------------------------------------------------------
__global__ __launch_bounds__(256) void wcvt_kernel(
    const float* __restrict__ wq, const float* __restrict__ wo,
    _Float16* __restrict__ wq16, _Float16* __restrict__ wo16) {
    const int NQ = 3 * C_DIM * C_DIM;                 // 196608
    int idx = (blockIdx.x * 256 + threadIdx.x) * 4;   // 65536 threads total
    if (idx < NQ) {
        f32x4 v = *(const f32x4*)(wq + idx);
        v4h h;
#pragma unroll
        for (int i = 0; i < 4; ++i) h[i] = (_Float16)v[i];
        *(v4h*)(wq16 + idx) = h;
    } else {
        int j = idx - NQ;                             // < 65536
        f32x4 v = *(const f32x4*)(wo + j);
        v4h h;
#pragma unroll
        for (int i = 0; i < 4; ++i) h[i] = (_Float16)v[i];
        *(v4h*)(wo16 + j) = h;
    }
}

// ---------------------------------------------------------------------------
// 1) GroupNorm -> xhT [B, L, C] f16 (spatial-major so QKV B-tiles are
//    contiguous). Each thread normalizes a 16-channel strip per position.
// ---------------------------------------------------------------------------
__global__ __launch_bounds__(256) void gn_kernel(
    const float* __restrict__ x, const float* __restrict__ gw,
    const float* __restrict__ gb, _Float16* __restrict__ xhT) {
    const int gid = blockIdx.x;            // 0..31
    const int b = gid / GROUPS, g = gid % GROUPS;
    const int gc0 = g * CPG;
    const size_t base = ((size_t)b * C_DIM + gc0) * L_DIM;
    const int tid = threadIdx.x;
    const int NELEM = CPG * L_DIM;         // 65536

    float s = 0.f, ss = 0.f;
    for (int i = tid; i < NELEM; i += 256) {
        float v = x[base + i];
        s += v; ss += v * v;
    }
    __shared__ float sh[256], sh2[256];
    sh[tid] = s; sh2[tid] = ss;
    __syncthreads();
    for (int off = 128; off > 0; off >>= 1) {
        if (tid < off) { sh[tid] += sh[tid + off]; sh2[tid] += sh2[tid + off]; }
        __syncthreads();
    }
    const float inv_n = 1.0f / (float)NELEM;
    const float mu = sh[0] * inv_n;
    const float var = sh2[0] * inv_n - mu * mu;
    const float rs = rsqrtf(var + 1e-5f);

    float wv[CPG], bv[CPG];
#pragma unroll
    for (int c = 0; c < CPG; ++c) { wv[c] = gw[gc0 + c] * rs; bv[c] = gb[gc0 + c]; }

    for (int i = tid; i < L_DIM; i += 256) {
        v16h row;
#pragma unroll
        for (int c = 0; c < CPG; ++c) {
            float v = (x[base + (size_t)c * L_DIM + i] - mu) * wv[c] + bv[c];
            row[c] = (_Float16)v;
        }
        *(v16h*)(&xhT[((size_t)b * L_DIM + i) * C_DIM + gc0]) = row;   // 32B
    }
}

// ---------------------------------------------------------------------------
// 2) QKV projection: one wave per 16x16 tile, K=256 in 8 WMMAs, all-f16
//    operands (2 x b128 each). head / q-k-v segment are tile-uniform.
// ---------------------------------------------------------------------------
__global__ __launch_bounds__(256) void qkv_kernel(
    const _Float16* __restrict__ wq16, const _Float16* __restrict__ xhT,
    _Float16* __restrict__ Qt, _Float16* __restrict__ Kt,
    _Float16* __restrict__ Vd) {
    const int lane = threadIdx.x, wave = threadIdx.y;
    const int tile = blockIdx.x * 8 + wave;          // B * 48 * 256 tiles
    const int b = tile / (48 * 256);
    const int rem = tile % (48 * 256);
    const int ot = rem / 256;                        // out-channel tile (768/16)
    const int nt = rem % 256;                        // spatial tile (4096/16)
    const int m = lane & 15, hi = lane >> 4, klo = hi * 16;

    const _Float16* wrow = wq16 + (size_t)(ot * 16 + m) * C_DIM;
    const _Float16* xrow = xhT + ((size_t)b * L_DIM + nt * 16 + m) * C_DIM;

    v8f acc = {};
    for (int k0 = 0; k0 < C_DIM; k0 += 32) {
        acc = wmma16(load_a16(wrow + k0, hi), load16h(xrow + k0 + klo), acc);
    }

    // tile-uniform decode (16 | 64 | 192 -> no tile straddles a boundary)
    const int obase = ot * 16;
    const int head = obase / 192;
    const int inner = obase - head * 192;
    const int which = inner >> 6;                    // 0=q 1=k 2=v
    const int dcb = inner & 63;                      // multiple of 16
    const int i = nt * 16 + m;
    const size_t hb = (size_t)b * NHEAD + head;

    v8h po;
#pragma unroll
    for (int r = 0; r < 8; ++r) po[r] = (_Float16)acc[r];

    if (which == 0) {
        *(v8h*)(&Qt[(hb * L_DIM + i) * HDIM + dcb + 8 * hi]) = po;     // 16B
    } else if (which == 1) {
        *(v8h*)(&Kt[(hb * L_DIM + i) * HDIM + dcb + 8 * hi]) = po;     // 16B
    } else {
#pragma unroll
        for (int r = 0; r < 8; ++r) {
            int dc = dcb + 8 * hi + r;
            Vd[(hb * HDIM + dc) * L_DIM + i] = po[r];   // dim-major for P*V^T
        }
    }
}

// ---------------------------------------------------------------------------
// 3) Flash attention. grid=(32, B*NHEAD), block=(32,8): one 16-query tile per
//    wave, 128 queries per block. Each 32-key block's K (32x64) and V (64x32,
//    dim-major) tiles are cooperatively staged in LDS (one coalesced 16B
//    global load per thread per buffer) and shared by all 8 waves. Per key
//    block: 4 WMMAs for S, online softmax (xor-shuffles within each 16-lane
//    half), P->A-layout via per-wave LDS (s_wait_dscnt), 4 WMMAs for O+=P*V^T.
//    O tile transposed through LDS -> OcT [B, L, C] f16.
// ---------------------------------------------------------------------------
__global__ __launch_bounds__(256) void attn_kernel(
    const _Float16* __restrict__ Qt, const _Float16* __restrict__ Kt,
    const _Float16* __restrict__ Vd, _Float16* __restrict__ OcT) {
    const int lane = threadIdx.x, wave = threadIdx.y;
    const int tid = wave * 32 + lane;
    const int bh = blockIdx.y;                        // b*NHEAD + h
    const int q0 = (blockIdx.x * 8 + wave) * 16;
    const int m = lane & 15, hi = lane >> 4, klo = hi * 16;

    __shared__ _Float16 Kblk[32 * 64];                // [key][dim]   4KB
    __shared__ _Float16 Vblk[64 * 32];                // [dim][key]   4KB
    __shared__ _Float16 pbuf[8][16 * 64];             // per-wave P / O staging
    _Float16* lds = pbuf[wave];

    const _Float16* Kg = Kt + (size_t)bh * L_DIM * HDIM;
    const _Float16* Vg = Vd + (size_t)bh * HDIM * L_DIM;

    // A = Q^T tile: M=query, K=head-dim (two 32-wide chunks), from global
    const _Float16* qrow = Qt + ((size_t)bh * L_DIM + q0 + m) * HDIM;
    v16h aq0 = load_a16(qrow, hi);        // head-dim 0..31
    v16h aq1 = load_a16(qrow + 32, hi);   // head-dim 32..63

    // staging assignments: K: 32 rows x 8 segs; V: 64 rows x 4 segs (16B each)
    const int krow_s = tid >> 3, kseg = tid & 7;      // 32 x 8 = 256
    const int vrow_s = tid >> 2, vseg = tid & 3;      // 64 x 4 = 256

    float mrow[8], lrow[8];
    v8f acc[4] = {};
#pragma unroll
    for (int r = 0; r < 8; ++r) { mrow[r] = -1e30f; lrow[r] = 0.f; }
    const float scale = 0.0625f;                      // 1/sqrt(C)

    for (int j0 = 0; j0 < L_DIM; j0 += 32) {
        __syncthreads();   // prior iteration's K/V reads complete (WAR)
        *(u32x4*)(Kblk + krow_s * HDIM + kseg * 8) =
            *(const u32x4*)(Kg + (size_t)(j0 + krow_s) * HDIM + kseg * 8);
        *(u32x4*)(Vblk + vrow_s * 32 + vseg * 8) =
            *(const u32x4*)(Vg + (size_t)vrow_s * L_DIM + j0 + vseg * 8);
        __syncthreads();   // staged tiles visible to all waves

        // ---- S = Q^T K for 32 keys (two 16-key sub-tiles) ----
        const _Float16* kr0 = Kblk + m * HDIM;          // key = m
        const _Float16* kr1 = Kblk + (16 + m) * HDIM;   // key = 16+m
        v8f s0 = {}, s1 = {};
        s0 = wmma16(aq0, load16h(kr0 + klo), s0);
        s0 = wmma16(aq1, load16h(kr0 + 32 + klo), s0);
        s1 = wmma16(aq0, load16h(kr1 + klo), s1);
        s1 = wmma16(aq1, load16h(kr1 + 32 + klo), s1);

        // ---- online softmax over this 32-key block ----
#pragma unroll
        for (int r = 0; r < 8; ++r) {
            float v0 = s0[r] * scale, v1 = s1[r] * scale;
            float mx = fmaxf(v0, v1);
#pragma unroll
            for (int off = 8; off >= 1; off >>= 1)
                mx = fmaxf(mx, __shfl_xor(mx, off, 32));   // within 16-lane half
            float mn = fmaxf(mrow[r], mx);
            float alpha = __expf(mrow[r] - mn);
            float p0 = __expf(v0 - mn), p1 = __expf(v1 - mn);
            float rs = p0 + p1;
#pragma unroll
            for (int off = 8; off >= 1; off >>= 1)
                rs += __shfl_xor(rs, off, 32);
            lrow[r] = lrow[r] * alpha + rs;
            mrow[r] = mn;
#pragma unroll
            for (int t = 0; t < 4; ++t) acc[t][r] *= alpha;
            // stage P (C-layout) into per-wave LDS row-major [16 x 32]
            lds[(r + 8 * hi) * 32 + m]      = (_Float16)p0;
            lds[(r + 8 * hi) * 32 + m + 16] = (_Float16)p1;
        }
        asm volatile("s_wait_dscnt 0" ::: "memory");   // cross-lane LDS RAW

        v16h ap = load_a16(lds + m * 32, hi);          // 2 x ds_load_b128

        // ---- O += P * V^T  (4 dim tiles of 16) ----
#pragma unroll
        for (int t = 0; t < 4; ++t) {
            const _Float16* vrow = Vblk + (t * 16 + m) * 32;   // dim = t*16+m
            acc[t] = wmma16(ap, load16h(vrow + klo), acc[t]);
        }
        asm volatile("s_wait_dscnt 0" ::: "memory");   // WAR before next P stores
    }

    // ---- normalize, transpose O tile through LDS, store OcT [B, L, C] ----
    const int b = bh >> 2, h = bh & 3;
#pragma unroll
    for (int t = 0; t < 4; ++t) {
#pragma unroll
        for (int r = 0; r < 8; ++r) {
            float o = acc[t][r] / lrow[r];
            lds[(r + 8 * hi) * 64 + t * 16 + m] = (_Float16)o;
        }
    }
    asm volatile("s_wait_dscnt 0" ::: "memory");
    // each lane emits one 32B contiguous run of OcT: row m, channel seg hi
    {
        const _Float16* src = lds + m * 64 + hi * 32;
        u32x4 d0 = *(const u32x4*)src;
        u32x4 d1 = *(const u32x4*)(src + 8);
        _Float16* dst = OcT + ((size_t)b * L_DIM + q0 + m) * C_DIM + h * HDIM + hi * 32;
        *(u32x4*)dst = d0;
        *(u32x4*)(dst + 8) = d1;
    }
}

// ---------------------------------------------------------------------------
// 4) Output projection + bias + residual (fp32 out), all-f16 WMMA operands.
// ---------------------------------------------------------------------------
__global__ __launch_bounds__(256) void proj_kernel(
    const _Float16* __restrict__ wo16, const float* __restrict__ obias,
    const _Float16* __restrict__ OcT, const float* __restrict__ xin,
    float* __restrict__ out) {
    const int lane = threadIdx.x, wave = threadIdx.y;
    const int tile = blockIdx.x * 8 + wave;           // B * 16 * 256 tiles
    const int b = tile / (16 * 256);
    const int rem = tile % (16 * 256);
    const int ct = rem / 256, nt = rem % 256;
    const int m = lane & 15, hi = lane >> 4, klo = hi * 16;

    const _Float16* wrow = wo16 + (size_t)(ct * 16 + m) * C_DIM;
    const _Float16* orow = OcT + ((size_t)b * L_DIM + nt * 16 + m) * C_DIM;

    v8f acc = {};
    for (int k0 = 0; k0 < C_DIM; k0 += 32) {
        acc = wmma16(load_a16(wrow + k0, hi), load16h(orow + k0 + klo), acc);
    }
#pragma unroll
    for (int r = 0; r < 8; ++r) {
        int c = ct * 16 + r + 8 * hi;
        size_t idx = ((size_t)b * C_DIM + c) * L_DIM + nt * 16 + m;
        out[idx] = acc[r] + obias[c] + xin[idx];
    }
}

// ---------------------------------------------------------------------------
extern "C" void kernel_launch(void* const* d_in, const int* in_sizes, int n_in,
                              void* d_out, int out_size, void* d_ws, size_t ws_size,
                              hipStream_t stream) {
    const float* x    = (const float*)d_in[0];   // [B, C, H, W]
    const float* gnw  = (const float*)d_in[1];   // [C]
    const float* gnb  = (const float*)d_in[2];   // [C]
    const float* wqkv = (const float*)d_in[3];   // [3C, C]
    const float* wout = (const float*)d_in[4];   // [C, C]
    const float* ob   = (const float*)d_in[5];   // [C]
    float* out = (float*)d_out;

    const size_t NE = (size_t)B_DIM * C_DIM * L_DIM;   // 2,097,152 elems
    _Float16* ws = (_Float16*)d_ws;
    _Float16* xhT  = ws;                 // [B, L, C]       f16
    _Float16* Qt   = ws + 1 * NE;        // [B*H, L, d]     f16
    _Float16* Kt   = ws + 2 * NE;        // [B*H, L, d]     f16
    _Float16* Vd   = ws + 3 * NE;        // [B*H, d, L]     f16
    _Float16* OcT  = ws + 4 * NE;        // [B, L, C]       f16
    _Float16* wq16 = ws + 5 * NE;        // [3C, C]         f16
    _Float16* wo16 = wq16 + (size_t)3 * C_DIM * C_DIM;   // [C, C] f16
    (void)ws_size; (void)in_sizes; (void)n_in; (void)out_size;

    wcvt_kernel<<<dim3(256), dim3(256), 0, stream>>>(wqkv, wout, wq16, wo16);
    gn_kernel<<<dim3(B_DIM * GROUPS), dim3(256), 0, stream>>>(x, gnw, gnb, xhT);

    // B * (768/16) * (4096/16) = 24576 tiles, 8 waves/block
    qkv_kernel<<<dim3(24576 / 8), dim3(32, 8), 0, stream>>>(wq16, xhT, Qt, Kt, Vd);

    // 128 queries per block; grid (4096/128, B*NHEAD)
    attn_kernel<<<dim3(32, B_DIM * NHEAD), dim3(32, 8), 0, stream>>>(Qt, Kt, Vd, OcT);

    // B * (256/16) * (4096/16) = 8192 tiles, 8 waves/block
    proj_kernel<<<dim3(8192 / 8), dim3(32, 8), 0, stream>>>(wo16, ob, OcT, x, out);
}
